// BondValencePredictor_85753317032080
// MI455X (gfx1250) — compile-verified, matrix operand order checked by reading, not run on hardware
//
#include <hip/hip_runtime.h>

// ---------------------------------------------------------------------------
// BondValencePredictor fused kernel for gfx1250 (MI455X, wave32, WMMA).
//
//   keys    = leaky(x @ Wk + bk)                    [64,128,256]
//   Q       = leaky(x @ Wq + bq)                    [64,128,6144] -> [B,A,H,4,6]
//   bdata[a,b,d,e,f] = sum_c keys[a,b,c]*Q[a,f,c,d,e];  mask f>b -> -inf
//
// Strategy: bf16 V_WMMA_F32_16X16X32_BF16 everywhere; fuse the Q GEMM with
// the einsum per (a, de=d*6+e) workgroup so the 201MB Q tensor never touches
// HBM. Q slice is written to LDS pre-packed in the exact WMMA B-fragment
// layout (lane/element mapping from CDNA5 ISA 16-bit B 32x16).
// ---------------------------------------------------------------------------

#define LEAK 0.1f

typedef __attribute__((ext_vector_type(16))) __bf16 bf16x16;
typedef __attribute__((ext_vector_type(8)))  __bf16 bf16x8;
typedef __attribute__((ext_vector_type(4)))  __bf16 bf16x4;
typedef __attribute__((ext_vector_type(8)))  float  f32x8;
typedef __attribute__((ext_vector_type(4)))  float  f32x4;

static __device__ __forceinline__ f32x8 wmma_bf16(bf16x16 a, bf16x16 b, f32x8 c) {
  return __builtin_amdgcn_wmma_f32_16x16x32_bf16(false, a, false, b, (short)0, c,
                                                 false, false);
}

// Load one 16x32 bf16 A-fragment (or 32x16 B-fragment, symmetric layout) from
// a row-major bf16 matrix. Per-lane K layout (ISA 7.12.2, 16-bit A 16x32):
//   lanes 0-15 : K = {0..7} u {16..23},  lanes 16-31 : K = {8..15} u {24..31}
// => per lane: two contiguous 8-element (16B) chunks at klo and klo+16.
static __device__ __forceinline__ bf16x16
load_frag(const __bf16* base, int row, int ld, int kbase, int lane) {
  const int klo = kbase + ((lane >> 4) << 3);
  const __bf16* p = base + (size_t)row * ld + klo;
  bf16x8 lo = *reinterpret_cast<const bf16x8*>(p);
  bf16x8 hi = *reinterpret_cast<const bf16x8*>(p + 16);
  bf16x16 r;
#pragma unroll
  for (int i = 0; i < 8; ++i) { r[i] = lo[i]; r[i + 8] = hi[i]; }
  return r;
}

// ---------------- stage 0: f32 -> bf16 convert (x), 4 elems/thread ----------
__global__ void cvt_bf16_kernel(const float* __restrict__ in,
                                __bf16* __restrict__ out, int n4) {
  int i = blockIdx.x * blockDim.x + threadIdx.x;
  if (i < n4) {
    f32x4 v = reinterpret_cast<const f32x4*>(in)[i];
    bf16x4 o;
    o.x = (__bf16)v.x; o.y = (__bf16)v.y; o.z = (__bf16)v.z; o.w = (__bf16)v.w;
    reinterpret_cast<bf16x4*>(out)[i] = o;
  }
}

// -------- stage 0b: transpose+convert W [K,cols] -> WT bf16 [cols,K] --------
__global__ void cvt_transpose_kernel(const float* __restrict__ in,
                                     __bf16* __restrict__ out,
                                     int rowsK, int cols) {
  int tid = blockIdx.x * blockDim.x + threadIdx.x;
  if (tid < rowsK * cols) {
    int col = tid / rowsK;
    int k   = tid - col * rowsK;
    out[tid] = (__bf16)in[(size_t)k * cols + col];
  }
}

// ---------------- stage 1: keys = leaky(x @ Wk + bk) in bf16 ----------------
// 8192x512x256 GEMM; one 16x16 tile per wave, 16 K-steps of wmma 16x16x32.
__global__ __launch_bounds__(256) void
keys_kernel(const __bf16* __restrict__ xbf, const __bf16* __restrict__ wkT,
            const float* __restrict__ bk, __bf16* __restrict__ keys) {
  const int lane = threadIdx.x & 31, w = threadIdx.x >> 5;
  const int t  = blockIdx.x * 8 + w;   // 8192 tiles total
  const int rt = t >> 4, ct = t & 15;
  const int row = rt * 16 + (lane & 15);
  const int c   = ct * 16 + (lane & 15);
  f32x8 acc = {};
#pragma unroll
  for (int kk = 0; kk < 16; ++kk) {
    bf16x16 aF = load_frag(xbf, row, 512, kk * 32, lane);
    bf16x16 bF = load_frag(wkT, c,   512, kk * 32, lane);
    acc = wmma_bf16(aF, bF, acc);
  }
  const float bias = bk[c];
#pragma unroll
  for (int r = 0; r < 8; ++r) {
    int rr = rt * 16 + r + ((lane >> 4) << 3);     // C/D layout: M = r + 8*hi
    float v = acc[r] + bias;
    v = (v > 0.0f) ? v : v * LEAK;
    keys[(size_t)rr * 256 + c] = (__bf16)v;
  }
}

// ---------------- stage 2: fused Q-GEMM + einsum + mask ---------------------
// One block per (a, de). 8 waves. LDS holds Q^T packed as B-fragments:
//   qfrag[cTile32(8)][fTile16(8)][lane(32)][elem(16)]  == 64 KB exactly.
__global__ __launch_bounds__(256) void
bond_main_kernel(const __bf16* __restrict__ xbf, const __bf16* __restrict__ wqT,
                 const float* __restrict__ bq, const __bf16* __restrict__ keys,
                 float* __restrict__ out) {
  __shared__ __bf16 qfrag[8 * 8 * 32 * 16];        // 65536 bytes
  const int lane = threadIdx.x & 31, w = threadIdx.x >> 5;
  const int a  = blockIdx.x / 24;
  const int de = blockIdx.x % 24;                  // d*6+e; Wq col = c*24 + de
  const __bf16* xa = xbf + (size_t)a * 128 * 512;

  // ---- Phase A: Q[f,c] = leaky(x[a] @ Wq[:,c*24+de] + bq) -> LDS (bf16) ----
  // Wave w owns f-rows [16w,16w+16); 16 c-tiles in two halves of 8 (VGPRs).
  const int fRow = 16 * w + (lane & 15);
#pragma unroll 1
  for (int half = 0; half < 2; ++half) {
    f32x8 acc[8] = {};
    for (int kk = 0; kk < 16; ++kk) {
      bf16x16 aF = load_frag(xa, fRow, 512, kk * 32, lane);
#pragma unroll
      for (int c8 = 0; c8 < 8; ++c8) {
        int col = (16 * (half * 8 + c8) + (lane & 15)) * 24 + de;
        bf16x16 bF = load_frag(wqT, col, 512, kk * 32, lane);
        acc[c8] = wmma_bf16(aF, bF, acc[c8]);
      }
    }
#pragma unroll
    for (int c8 = 0; c8 < 8; ++c8) {
      int c = 16 * (half * 8 + c8) + (lane & 15);
      float bias = bq[c * 24 + de];
#pragma unroll
      for (int r = 0; r < 8; ++r) {
        int fl = r + ((lane >> 4) << 3);           // local f within 16-strip
        float v = acc[c8][r] + bias;
        v = (v > 0.0f) ? v : v * LEAK;
        // scatter into B-fragment layout: k = c&31 selects lane-half/elem
        int k = c & 31;
        int dlane = fl | (((k >> 3) & 1) << 4);
        int di    = ((k >> 4) << 3) | (k & 7);
        qfrag[(((c >> 5) * 8 + w) * 32 + dlane) * 16 + di] = (__bf16)v;
      }
    }
  }
  __syncthreads();

  // ---- Phase B: O[b,f] = keys[a] @ Q^T, mask f>b, write f32 ----------------
  const __bf16* ka = keys + (size_t)a * 128 * 256;
  const int bRow = 16 * w + (lane & 15);
  bf16x16 aK[8];
#pragma unroll
  for (int ck = 0; ck < 8; ++ck) aK[ck] = load_frag(ka, bRow, 256, ck * 32, lane);
#pragma unroll 1
  for (int ft = 0; ft < 8; ++ft) {
    f32x8 acc = {};
#pragma unroll
    for (int ck = 0; ck < 8; ++ck) {
      // B-fragment is contiguous 32B per lane in LDS (two 16B ds loads)
      const bf16x8* q =
          reinterpret_cast<const bf16x8*>(&qfrag[((ck * 8 + ft) * 32 + lane) * 16]);
      bf16x8 lo = q[0], hi = q[1];
      bf16x16 bF;
#pragma unroll
      for (int i = 0; i < 8; ++i) { bF[i] = lo[i]; bF[i + 8] = hi[i]; }
      acc = wmma_bf16(aK[ck], bF, acc);
    }
    const int f = 16 * ft + (lane & 15);
    float* outBase = out + ((size_t)(a * 128) * 24 + de) * 128 + f;
#pragma unroll
    for (int r = 0; r < 8; ++r) {
      int b = 16 * w + r + ((lane >> 4) << 3);
      float v = (f > b) ? -__builtin_inff() : acc[r];
      outBase[(size_t)b * 24 * 128] = v;
    }
  }
}

// ---------------------------------------------------------------------------
extern "C" void kernel_launch(void* const* d_in, const int* in_sizes, int n_in,
                              void* d_out, int out_size, void* d_ws, size_t ws_size,
                              hipStream_t stream) {
  const float* x  = (const float*)d_in[0];   // [64,128,512]
  const float* Wk = (const float*)d_in[1];   // [512,256]
  const float* bk = (const float*)d_in[2];   // [256]
  const float* Wq = (const float*)d_in[3];   // [512,6144]
  const float* bq = (const float*)d_in[4];   // [6144]
  float* out = (float*)d_out;                // [64,128,4,6,128]

  char* ws = (char*)d_ws;                    // ~19.1 MB used
  __bf16* xbf  = (__bf16*)(ws);                                  // 8,388,608 B
  __bf16* wqT  = (__bf16*)(ws + 8388608);                        // 6,291,456 B
  __bf16* wkT  = (__bf16*)(ws + 8388608 + 6291456);              //   262,144 B
  __bf16* keys = (__bf16*)(ws + 8388608 + 6291456 + 262144);     // 4,194,304 B

  cvt_bf16_kernel<<<4096, 256, 0, stream>>>(x, xbf, 1048576);        // 64*128*512/4
  cvt_transpose_kernel<<<12288, 256, 0, stream>>>(Wq, wqT, 512, 6144);
  cvt_transpose_kernel<<<512,   256, 0, stream>>>(Wk, wkT, 512, 256);
  keys_kernel<<<1024, 256, 0, stream>>>(xbf, wkT, bk, keys);
  bond_main_kernel<<<64 * 24, 256, 0, stream>>>(xbf, wqT, bq, keys, out);
}
